// AttnAllLayer_20882130993411
// MI455X (gfx1250) — matrix operand order repaired and checked by reading, not compiled, not run on hardware
//
#include <hip/hip_runtime.h>

typedef unsigned short u16;
typedef __attribute__((ext_vector_type(16))) __bf16 v16bf;
typedef __attribute__((ext_vector_type(8))) float v8f;

static __device__ inline u16 f2bf(float f) {
  unsigned u = __builtin_bit_cast(unsigned, f);
  unsigned r = (u + 0x7FFFu + ((u >> 16) & 1u)) >> 16;
  return (u16)r;
}

// ---------------------------------------------------------------------------
// Generic batched bf16 WMMA GEMM: C(f32, MxN) = A(bf16, MxK) * B(bf16, KxN)
// grid.x = ceil(M/64)*ceil(N/32), grid.y = 96 (= L*B); block 256 = 8 waves.
// K multiple of 32; M,N multiples of 16 (edge rows/cols clamped: pad results
// are computed but never stored, so no zero-fill needed along M/N; K pads are
// zeroed by the converter kernels).
// A panel staged with async global->LDS (ASYNCcnt); B staged manually into
// WMMA fragment order so fragments load as 2x ds_load_b128 per operand.
// ---------------------------------------------------------------------------
__global__ __launch_bounds__(256)
void k_gemm_bf16(const u16* __restrict__ A, long a_ls, long a_bs, int lda,
                 const u16* __restrict__ B, long b_ls, long b_bs, int ldb,
                 float* __restrict__ C, long c_ls, long c_bs, int ldc,
                 int M, int N, int K)
{
  int lb = blockIdx.y; int l = lb >> 5; int b = lb & 31;
  A += (long)l * a_ls + (long)b * a_bs;
  B += (long)l * b_ls + (long)b * b_bs;
  C += (long)l * c_ls + (long)b * c_bs;
  int nbn = (N + 31) / 32;
  int bm = blockIdx.x / nbn, bn = blockIdx.x % nbn;
  int m0 = bm * 64, n0 = bn * 32;
  // fragment-order panels: [tile][lane][16 slots]
  __shared__ u16 As[4 * 32 * 16];   // 64x32 panel
  __shared__ u16 Bs[2 * 32 * 16];   // 32x32 panel
  int tid = threadIdx.x;
  int wid = tid >> 5, lane = tid & 31;
  int wm = wid & 3, wn = wid >> 2;            // wave tile: rows m0+wm*16, cols n0+wn*16
  int tm = m0 + wm * 16, tn = n0 + wn * 16;

  // A staging map: one 16-byte async copy per thread per K-step
  int ar = tid >> 2;                 // 0..63 panel row
  int aseg = tid & 3;                // 8-element K segment
  int ac0 = aseg * 8;
  int agr = m0 + ar; if (agr > M - 1) agr = M - 1;   // clamp (pad rows unused)
  int alane = (ar & 15) + ((aseg & 1) ? 16 : 0);
  int ajb = (aseg >= 2) ? 8 : 0;
  u16* adst = &As[(((ar >> 4) * 32 + alane) << 4) + ajb];
  unsigned lds_a = (unsigned)(size_t)adst;            // low 32 bits = LDS offset
  const u16* arow = A + agr * lda + ac0;

  // B staging map: 4 consecutive K rows at one column per thread
  int bn_ = tid & 31;                // 0..31 panel col
  int br0 = (tid >> 5) * 4;          // 0,4,...,28 panel K row
  int bgc = n0 + bn_; if (bgc > N - 1) bgc = N - 1;  // clamp (pad cols unused)
  int blane = (bn_ & 15) + ((br0 & 8) ? 16 : 0);
  u16* bdst = &Bs[((((bn_ >> 4) * 32 + blane)) << 4) + (br0 & 7) + ((br0 >= 16) ? 8 : 0)];

  v8f acc = {};
  for (int k0 = 0; k0 < K; k0 += 32) {
    // stage A: async 16B global -> LDS (fragment-order address)
    {
      unsigned long long ga = (unsigned long long)(arow + k0);
      asm volatile("global_load_async_to_lds_b128 %0, %1, off"
                   :: "v"(lds_a), "v"(ga) : "memory");
    }
    // stage B: 4 coalesced u16 rows -> one 8B LDS store
    {
      union { u16 h[4]; uint2 d; } pk;
      const u16* bp = B + (k0 + br0) * ldb + bgc;
      pk.h[0] = bp[0];
      pk.h[1] = bp[ldb];
      pk.h[2] = bp[2 * ldb];
      pk.h[3] = bp[3 * ldb];
      *(uint2*)bdst = pk.d;
    }
    asm volatile("s_wait_asynccnt 0" ::: "memory");
    __syncthreads();
    // fragments: contiguous 32B per lane -> 2x b128 LDS loads each
    union { uint4 q[2]; v16bf v; } af, bf;
    {
      const uint4* pa = (const uint4*)&As[((wm * 32 + lane) << 4)];
      af.q[0] = pa[0]; af.q[1] = pa[1];
      const uint4* pb = (const uint4*)&Bs[((wn * 32 + lane) << 4)];
      bf.q[0] = pb[0]; bf.q[1] = pb[1];
    }
    acc = __builtin_amdgcn_wmma_f32_16x16x32_bf16(false, af.v, false, bf.v,
                                                  (short)0, acc, false, false);
    __syncthreads();
  }
  // store D: VGPR r -> row tm + r (+8 for upper lanes), col tn + lane%16
  {
    int ncol = lane & 15;
    int mh = (lane < 16) ? 0 : 8;
#pragma unroll
    for (int r = 0; r < 8; ++r) {
      int gm = tm + mh + r, gn = tn + ncol;
      if (gm < M && gn < N) C[(long)gm * ldc + gn] = acc[r];
    }
  }
}

// ---------------------------------------------------------------------------
// f32 -> bf16 converters with zero padding, optional per-row/col bias.
// grid.x = ceil(R*Cc/256), grid.y = L*b_per_l
// ---------------------------------------------------------------------------
__global__ void k_cvt_nt(u16* dst, long d_ls, long d_bs,
                         const float* src, long s_ls, long s_bs, int s_ld,
                         const float* bias, long bias_ls,
                         int R, int Cc, int vr, int vc, int b_per_l)
{
  int bz = blockIdx.y, l = bz / b_per_l, b = bz % b_per_l;
  dst += (long)l * d_ls + (long)b * d_bs;
  src += (long)l * s_ls + (long)b * s_bs;
  const float* bi = bias ? bias + (long)l * bias_ls : nullptr;
  int i = blockIdx.x * 256 + threadIdx.x;
  if (i >= R * Cc) return;
  int r = i / Cc, c = i % Cc;
  u16 o = 0;
  if (r < vr && c < vc) {
    float v = src[(long)r * s_ld + c];
    if (bi) v += bi[r];
    o = f2bf(v);
  }
  dst[i] = o;
}

__global__ void k_cvt_t(u16* dst, long d_ls, long d_bs,
                        const float* src, long s_ls, long s_bs, int s_ld,
                        const float* bias, long bias_ls,
                        int R, int Cc, int vr, int vc, int b_per_l)
{
  int bz = blockIdx.y, l = bz / b_per_l, b = bz % b_per_l;
  dst += (long)l * d_ls + (long)b * d_bs;
  src += (long)l * s_ls + (long)b * s_bs;
  const float* bi = bias ? bias + (long)l * bias_ls : nullptr;
  int i = blockIdx.x * 256 + threadIdx.x;
  if (i >= R * Cc) return;
  int r = i / Cc, c = i % Cc;
  u16 o = 0;
  if (r < vr && c < vc) {
    float v = src[(long)c * s_ld + r];
    if (bi) v += bi[c];
    o = f2bf(v);
  }
  dst[i] = o;
}

// ---------------------------------------------------------------------------
// Mask branch: c1 = relu(conv3x3 valid, [mask,1-mask] -> 4ch), 125x125
// ---------------------------------------------------------------------------
__global__ void k_c1(float* __restrict__ c1, const float* __restrict__ mask,
                     const float* __restrict__ w1, const float* __restrict__ b1)
{
  int lb = blockIdx.y; int l = lb >> 5, b = lb & 31;
  float* out = c1 + (long)lb * 62500;
  const float* wk = w1 + (long)l * 4 * 2 * 9;
  const float* bk = b1 + (long)l * 4;
  const float* mk = mask + (long)b * 127 * 127;
  int i = blockIdx.x * 256 + threadIdx.x;
  if (i >= 4 * 125 * 125) return;
  int oc = i / 15625, rem = i % 15625, y = rem / 125, x = rem % 125;
  float s = bk[oc];
  for (int dy = 0; dy < 3; ++dy)
    for (int dx = 0; dx < 3; ++dx) {
      float mv = mk[(y + dy) * 127 + x + dx];
      s += wk[oc * 18 + dy * 3 + dx] * mv + wk[oc * 18 + 9 + dy * 3 + dx] * (1.f - mv);
    }
  out[i] = fmaxf(s, 0.f);
}

// rest of mask chain in one block per (l,b): c2@resize -> c3 SAME -> resize -> c4 SAME -> m(15x15)
__global__ __launch_bounds__(256)
void k_mrest(float* __restrict__ mW, const float* __restrict__ c1buf,
             const float* __restrict__ w2, const float* __restrict__ b2,
             const float* __restrict__ w3, const float* __restrict__ b3,
             const float* __restrict__ w4, const float* __restrict__ b4)
{
  int lb = blockIdx.x; int l = lb >> 5;
  const float* c1 = c1buf + (long)lb * 62500;
  const float* W2 = w2 + (long)l * 8 * 4 * 9; const float* B2 = b2 + (long)l * 8;
  const float* W3 = w3 + (long)l * 4 * 8 * 9; const float* B3 = b3 + (long)l * 4;
  const float* W4 = w4 + (long)l * 4 * 9;     const float  B4 = b4[l];
  __shared__ float r2[8 * 196];
  __shared__ float c3s[4 * 196];
  __shared__ float r3[4 * 225];
  int tid = threadIdx.x;
  for (int i = tid; i < 8 * 196; i += 256) {
    int oc = i / 196, rem = i % 196, j = rem / 14, q = rem % 14;
    float sy = 122.f / 13.f;
    float fy = j * sy; int y0 = (int)floorf(fy); y0 = y0 < 0 ? 0 : (y0 > 121 ? 121 : y0);
    float wy = fy - y0;
    float fx = q * sy; int x0 = (int)floorf(fx); x0 = x0 < 0 ? 0 : (x0 > 121 ? 121 : x0);
    float wx = fx - x0;
    float acc = 0.f;
    for (int d2 = 0; d2 < 4; ++d2) {
      int cy = y0 + (d2 >> 1), cx = x0 + (d2 & 1);
      float s = B2[oc];
      for (int ic = 0; ic < 4; ++ic) {
        const float* cp = c1 + ic * 15625 + cy * 125 + cx;
        const float* wk = W2 + (oc * 4 + ic) * 9;
        s += wk[0]*cp[0]   + wk[1]*cp[1]   + wk[2]*cp[2]
           + wk[3]*cp[125] + wk[4]*cp[126] + wk[5]*cp[127]
           + wk[6]*cp[250] + wk[7]*cp[251] + wk[8]*cp[252];
      }
      s = fmaxf(s, 0.f);
      acc += s * ((d2 >> 1) ? wy : 1.f - wy) * ((d2 & 1) ? wx : 1.f - wx);
    }
    r2[i] = acc;
  }
  __syncthreads();
  for (int i = tid; i < 4 * 196; i += 256) {   // c3 = conv3x3 SAME 8->4 (no relu)
    int oc = i / 196, rem = i % 196, y = rem / 14, x = rem % 14;
    float s = B3[oc];
    for (int ic = 0; ic < 8; ++ic) {
      const float* rp = r2 + ic * 196;
      const float* wk = W3 + (oc * 8 + ic) * 9;
      for (int dy = -1; dy <= 1; ++dy)
        for (int dx = -1; dx <= 1; ++dx) {
          int yy = y + dy, xx = x + dx;
          if (yy >= 0 && yy < 14 && xx >= 0 && xx < 14)
            s += wk[(dy + 1) * 3 + dx + 1] * rp[yy * 14 + xx];
        }
    }
    c3s[i] = s;
  }
  __syncthreads();
  for (int i = tid; i < 4 * 225; i += 256) {   // resize 14 -> 15
    int oc = i / 225, rem = i % 225, j = rem / 15, q = rem % 15;
    float sy = 13.f / 14.f;
    float fy = j * sy; int y0 = (int)floorf(fy); y0 = y0 < 0 ? 0 : (y0 > 12 ? 12 : y0);
    float wy = fy - y0;
    float fx = q * sy; int x0 = (int)floorf(fx); x0 = x0 < 0 ? 0 : (x0 > 12 ? 12 : x0);
    float wx = fx - x0;
    const float* cp = c3s + oc * 196;
    float v00 = cp[y0 * 14 + x0], v01 = cp[y0 * 14 + x0 + 1];
    float v10 = cp[(y0 + 1) * 14 + x0], v11 = cp[(y0 + 1) * 14 + x0 + 1];
    r3[i] = v00 * (1 - wy) * (1 - wx) + v10 * wy * (1 - wx) + v01 * (1 - wy) * wx + v11 * wy * wx;
  }
  __syncthreads();
  for (int i = tid; i < 225; i += 256) {       // c4 = conv3x3 SAME 4->1 -> m
    int y = i / 15, x = i % 15;
    float s = B4;
    for (int ic = 0; ic < 4; ++ic) {
      const float* rp = r3 + ic * 225;
      const float* wk = W4 + ic * 9;
      for (int dy = -1; dy <= 1; ++dy)
        for (int dx = -1; dx <= 1; ++dx) {
          int yy = y + dy, xx = x + dx;
          if (yy >= 0 && yy < 15 && xx >= 0 && xx < 15)
            s += wk[(dy + 1) * 3 + dx + 1] * rp[yy * 15 + xx];
        }
    }
    mW[(long)lb * 225 + i] = s;
  }
}

// ---------------------------------------------------------------------------
// Precise RoI pooling (separable hat-window integral)
// ---------------------------------------------------------------------------
static __device__ inline float hat_int(float u) {
  if (u <= -1.f) return 0.f;
  if (u <= 0.f) { float t = u + 1.f; return 0.5f * t * t; }
  if (u <= 1.f) { float t = 1.f - u; return 1.f - 0.5f * t * t; }
  return 1.f;
}

__global__ __launch_bounds__(256)
void k_prroi(float* __restrict__ rois, const float* __restrict__ zf,
             const float* __restrict__ box)
{
  int lb = blockIdx.x; int b = lb & 31;
  __shared__ float Wy[105], Wx[105];
  const float SC = 15.f / 127.f;
  const float* bx = box + b * 4;
  float x1 = bx[0] * SC, y1 = bx[1] * SC, x2 = bx[2] * SC, y2 = bx[3] * SC;
  float bh = (y2 - y1) / 7.f, bw = (x2 - x1) / 7.f;
  int tid = threadIdx.x;
  if (tid < 105) {
    int p = tid / 15, h = tid % 15;
    Wy[tid] = hat_int(y1 + (p + 1) * bh - h) - hat_int(y1 + p * bh - h);
  } else if (tid < 210) {
    int t = tid - 105, q = t / 15, w = t % 15;
    Wx[t] = hat_int(x1 + (q + 1) * bw - w) - hat_int(x1 + q * bw - w);
  }
  __syncthreads();
  float inv_area = 1.f / fmaxf(bh * bw, 1e-6f);
  int c = tid;
  const float* zfc = zf + ((long)lb * 256 + c) * 225;
  for (int p = 0; p < 7; ++p) {
    float row[15];
    for (int w = 0; w < 15; ++w) row[w] = 0.f;
    for (int h = 0; h < 15; ++h) {
      float wv = Wy[p * 15 + h];
      for (int w = 0; w < 15; ++w) row[w] += wv * zfc[h * 15 + w];
    }
    for (int q = 0; q < 7; ++q) {
      float s = 0.f;
      for (int w = 0; w < 15; ++w) s += Wx[q * 15 + w] * row[w];
      rois[((long)lb * 256 + c) * 49 + p * 7 + q] = s * inv_area;
    }
  }
}

// ---------------------------------------------------------------------------
// Softmax row of S (240x240, valid 225x225), /sqrt(C)=16, write bf16 P (240x256)
// ---------------------------------------------------------------------------
__global__ __launch_bounds__(256)
void k_softmax(u16* __restrict__ P, const float* __restrict__ S)
{
  int lb = blockIdx.y, row = blockIdx.x, tid = threadIdx.x;
  u16* p = P + (long)lb * (240 * 256) + (long)row * 256;
  if (row >= 225) { p[tid] = 0; return; }
  const float* s = S + (long)lb * (240 * 240) + (long)row * 240;
  __shared__ float red[256];
  float x = (tid < 225) ? s[tid] : -3.4e38f;
  red[tid] = x; __syncthreads();
  for (int st = 128; st > 0; st >>= 1) {
    if (tid < st) red[tid] = fmaxf(red[tid], red[tid + st]);
    __syncthreads();
  }
  float mx = red[0]; __syncthreads();
  float ex = (tid < 225) ? __expf(x - mx) : 0.f;
  red[tid] = ex; __syncthreads();
  for (int st = 128; st > 0; st >>= 1) {
    if (tid < st) red[tid] += red[tid + st];
    __syncthreads();
  }
  float scale = 1.f / (red[0] * 16.f);
  p[tid] = (tid < 225) ? f2bf(ex * scale) : (u16)0;
}

// ---------------------------------------------------------------------------
// z_avg / z_max pooling + two MLPs (256->128 relu ->256)
// ---------------------------------------------------------------------------
__global__ __launch_bounds__(256)
void k_pool_mlp(float* __restrict__ zavg, float* __restrict__ zmax,
                const float* __restrict__ rois,
                const float* aw1, const float* ab1, const float* aw2, const float* ab2,
                const float* mw1, const float* mb1, const float* mw2, const float* mb2)
{
  int lb = blockIdx.x; int l = lb >> 5;
  __shared__ float pm[256], px[256], hid[256];
  int c = threadIdx.x;
  const float* r = rois + ((long)lb * 256 + c) * 49;
  float s = 0.f, mx = -3.4e38f;
  for (int i = 0; i < 49; ++i) { float v = r[i]; s += v; mx = fmaxf(mx, v); }
  pm[c] = s / 49.f; px[c] = mx;
  __syncthreads();
  if (c < 128) {
    float h = ab1[(long)l * 128 + c];
    const float* w = aw1 + ((long)l * 128 + c) * 256;
    for (int k = 0; k < 256; ++k) h += w[k] * pm[k];
    hid[c] = fmaxf(h, 0.f);
  } else {
    int j = c - 128;
    float h = mb1[(long)l * 128 + j];
    const float* w = mw1 + ((long)l * 128 + j) * 256;
    for (int k = 0; k < 256; ++k) h += w[k] * px[k];
    hid[c] = fmaxf(h, 0.f);
  }
  __syncthreads();
  float za = ab2[(long)l * 256 + c];
  const float* w2a = aw2 + ((long)l * 256 + c) * 128;
  for (int k = 0; k < 128; ++k) za += w2a[k] * hid[k];
  float zm = mb2[(long)l * 256 + c];
  const float* w2m = mw2 + ((long)l * 256 + c) * 128;
  for (int k = 0; k < 128; ++k) zm += w2m[k] * hid[128 + k];
  zavg[(long)lb * 256 + c] = za;
  zmax[(long)lb * 256 + c] = zm;
}

// ---------------------------------------------------------------------------
// chan_z tail (grouped einsum + relu + aw GEMV) and cue fusion (sigmoid)
// ---------------------------------------------------------------------------
__global__ __launch_bounds__(256)
void k_chanz_tail(float* __restrict__ cues, const float* __restrict__ fmat,
                  const float* __restrict__ zavg, const float* __restrict__ zmax,
                  const float* gw, const float* gb, const float* aw, const float* ab,
                  const float* drw, const float* drb)
{
  int lb = blockIdx.x; int l = lb >> 5;
  __shared__ float o2[512];
  int tid = threadIdx.x;
  for (int o = tid; o < 512; o += 256) {
    int g = o >> 1;
    const float* w = gw + ((long)l * 512 + o) * 256;
    const float* f = fmat + ((long)lb * 256 + g) * 256;
    float s = gb[(long)l * 512 + o];
    for (int h = 0; h < 256; ++h) s += w[h] * f[h];
    o2[o] = fmaxf(s, 0.f);
  }
  __syncthreads();
  int c = tid;
  float s = ab[(long)l * 256 + c];
  const float* wr = aw + ((long)l * 256 + c) * 512;
  for (int o = 0; o < 512; ++o) s += wr[o] * o2[o];
  const float* dw = drw + ((long)l * 256 + c) * 3;
  float za = zavg[(long)lb * 256 + c], zm = zmax[(long)lb * 256 + c];
  float t = dw[0] * za + dw[1] * zm + dw[2] * s + drb[(long)l * 256 + c];
  cues[(long)lb * 256 + c] = 1.f / (1.f + __expf(-t));
}

// ---------------------------------------------------------------------------
// sc branch collapse: v = zf @ m ; w = phi_w @ v + phi_b*sum(m)
//                     u = delta_w^T @ w ; kdot = delta_b . w
// ---------------------------------------------------------------------------
__global__ __launch_bounds__(256)
void k_sc_prep(float* __restrict__ uW, float* __restrict__ kdotW,
               const float* __restrict__ mW, const float* __restrict__ zf,
               const float* phiw, const float* phib,
               const float* dltw, const float* dltb)
{
  int lb = blockIdx.x; int l = lb >> 5;
  __shared__ float vS[256], wS[256], red[256];
  int c = threadIdx.x;
  const float* mp = mW + (long)lb * 225;
  const float* zfc = zf + ((long)lb * 256 + c) * 225;
  float v = 0.f, ms = 0.f;
  for (int i = 0; i < 225; ++i) { v += zfc[i] * mp[i]; ms += mp[i]; }
  vS[c] = v;
  __syncthreads();
  const float* pw = phiw + ((long)l * 256 + c) * 256;
  float w = phib[(long)l * 256 + c] * ms;
  for (int k = 0; k < 256; ++k) w += pw[k] * vS[k];
  wS[c] = w;
  __syncthreads();
  const float* dwb = dltw + (long)l * 65536;
  float u = 0.f;
  for (int k = 0; k < 256; ++k) u += dwb[(long)k * 256 + c] * wS[k];
  uW[(long)lb * 256 + c] = u;
  red[c] = dltb[(long)l * 256 + c] * wS[c];
  __syncthreads();
  for (int st = 128; st > 0; st >>= 1) {
    if (c < st) red[c] += red[c + st];
    __syncthreads();
  }
  if (c == 0) kdotW[lb] = red[0];
}

__global__ void k_sc(float* __restrict__ scW, const float* __restrict__ xf,
                     const float* __restrict__ uW, const float* __restrict__ kdotW)
{
  int lb = blockIdx.y;
  int x = blockIdx.x * 256 + threadIdx.x;
  if (x >= 961) return;
  const float* xfb = xf + (long)lb * 256 * 961;
  const float* u = uW + (long)lb * 256;
  float s = kdotW[lb];
  for (int c = 0; c < 256; ++c) s += xfb[(long)c * 961 + x] * u[c];
  s *= 0.0625f;                         // 1/sqrt(256)
  scW[(long)lb * 961 + x] = fminf(fmaxf(s, 0.f), 1.f);
}

// ---------------------------------------------------------------------------
// Output kernels
// ---------------------------------------------------------------------------
__global__ void k_xf_out(float* __restrict__ out, const float* __restrict__ xf,
                         const float* __restrict__ cues, const float* __restrict__ scW,
                         long O2)
{
  int lb = blockIdx.y;
  int i = blockIdx.x * 256 + threadIdx.x;
  if (i >= 256 * 961) return;
  int c = i / 961, x = i % 961;
  float v = xf[(long)lb * 246016 + i];
  out[O2 + (long)lb * 246016 + i] =
      v * (1.f + cues[(long)lb * 256 + c]) - scW[(long)lb * 961 + x];
}

__global__ void k_zf_out(float* __restrict__ out, const float* __restrict__ zf,
                         const float* __restrict__ SA, const float* __restrict__ cues,
                         const float* bng, const float* bnb, const float* bnm,
                         const float* bnv, long O0, long O1)
{
  int lb = blockIdx.y; int l = lb >> 5;
  int i = blockIdx.x * 256 + threadIdx.x;
  if (i >= 256 * 225) return;
  int c = i / 225, px = i % 225;
  float z = zf[(long)lb * 57600 + i];
  float sa = SA[(long)lb * 61440 + (long)c * 240 + px];
  float inv = rsqrtf(bnv[(long)l * 256 + c] + 1e-5f);
  float sab = (sa - bnm[(long)l * 256 + c]) * inv * bng[(long)l * 256 + c]
              + bnb[(long)l * 256 + c];
  float full = z * (1.f + cues[(long)lb * 256 + c]) + sab;
  out[O1 + (long)lb * 57600 + i] = full;
  int y = px / 15, x = px % 15;
  if (y >= 4 && y < 11 && x >= 4 && x < 11)
    out[O0 + ((long)lb * 256 + c) * 49 + (y - 4) * 7 + (x - 4)] = full;
}

// ===========================================================================
// Host-side orchestration
// ===========================================================================
static inline int cdiv(long a, long b) { return (int)((a + b - 1) / b); }

extern "C" void kernel_launch(void* const* d_in, const int* in_sizes, int n_in,
                              void* d_out, int out_size, void* d_ws, size_t ws_size,
                              hipStream_t stream) {
  (void)in_sizes; (void)n_in; (void)out_size;
  const float* zf     = (const float*)d_in[0];
  const float* xf     = (const float*)d_in[1];
  const float* mask   = (const float*)d_in[2];
  const float* box    = (const float*)d_in[3];
  const float* nl_g_w = (const float*)d_in[4];
  const float* nl_g_b = (const float*)d_in[5];
  const float* nl_th_w= (const float*)d_in[6];
  const float* nl_th_b= (const float*)d_in[7];
  const float* nl_ph_w= (const float*)d_in[8];
  const float* nl_ph_b= (const float*)d_in[9];
  const float* nl_out_w=(const float*)d_in[10];
  const float* bn_g   = (const float*)d_in[11];
  const float* bn_b   = (const float*)d_in[12];
  const float* bn_m   = (const float*)d_in[13];
  const float* bn_v   = (const float*)d_in[14];
  const float* cz_th_w= (const float*)d_in[15];
  const float* cz_th_b= (const float*)d_in[16];
  const float* cz_ph_w= (const float*)d_in[17];
  const float* cz_ph_b= (const float*)d_in[18];
  const float* cz_gw  = (const float*)d_in[19];
  const float* cz_gb  = (const float*)d_in[20];
  const float* cz_aw  = (const float*)d_in[21];
  const float* cz_ab  = (const float*)d_in[22];
  const float* ap_w1  = (const float*)d_in[23];
  const float* ap_b1  = (const float*)d_in[24];
  const float* ap_w2  = (const float*)d_in[25];
  const float* ap_b2  = (const float*)d_in[26];
  const float* mp_w1  = (const float*)d_in[27];
  const float* mp_b1  = (const float*)d_in[28];
  const float* mp_w2  = (const float*)d_in[29];
  const float* mp_b2  = (const float*)d_in[30];
  const float* dr_w   = (const float*)d_in[31];
  const float* dr_b   = (const float*)d_in[32];
  const float* ac1_w  = (const float*)d_in[33];
  const float* ac1_b  = (const float*)d_in[34];
  const float* ac2_w  = (const float*)d_in[35];
  const float* ac2_b  = (const float*)d_in[36];
  const float* ac3_w  = (const float*)d_in[37];
  const float* ac3_b  = (const float*)d_in[38];
  const float* ac4_w  = (const float*)d_in[39];
  const float* ac4_b  = (const float*)d_in[40];
  const float* phi_w  = (const float*)d_in[41];
  const float* phi_b  = (const float*)d_in[42];
  const float* delta_w= (const float*)d_in[43];
  const float* delta_b= (const float*)d_in[44];
  float* out = (float*)d_out;

  const long LB = 96;
  size_t off = 0;
  auto alloc = [&](size_t bytes) { size_t o = off; off += (bytes + 255) & ~size_t(255); return o; };
  char* W = (char*)d_ws;
  size_t o_wcat  = alloc(3L * 384 * 256 * 2);
  size_t o_wout  = alloc(3L * 256 * 128 * 2);
  size_t o_wcz   = alloc(3L * 512 * 256 * 2);
  size_t o_c1    = alloc(LB * 62500 * 4);
  size_t o_m     = alloc(LB * 225 * 4);
  size_t o_rois  = alloc(LB * 12544 * 4);
  size_t o_roisbf= alloc(LB * 16384 * 2);
  size_t o_zfbf  = alloc(LB * 61440 * 2);
  size_t o_gtp   = alloc(LB * 92160 * 4);
  size_t o_thT   = alloc(LB * 30720 * 2);
  size_t o_phb   = alloc(LB * 30720 * 2);
  size_t o_gT    = alloc(LB * 32768 * 2);
  size_t o_S     = alloc(LB * 57600 * 4);
  size_t o_Pbf   = alloc(LB * 61440 * 2);
  size_t o_Yp    = alloc(LB * 30720 * 4);
  size_t o_Ybf   = alloc(LB * 30720 * 2);
  size_t o_SA    = alloc(LB * 61440 * 4);
  size_t o_thph  = alloc(LB * 32768 * 4);
  size_t o_thbf  = alloc(LB * 16384 * 2);
  size_t o_phTbf = alloc(LB * 16384 * 2);
  size_t o_fmat  = alloc(LB * 65536 * 4);
  size_t o_zavg  = alloc(LB * 256 * 4);
  size_t o_zmax  = alloc(LB * 256 * 4);
  size_t o_cues  = alloc(LB * 256 * 4);
  size_t o_u     = alloc(LB * 256 * 4);
  size_t o_kdot  = alloc(LB * 4);
  size_t o_sc    = alloc(LB * 961 * 4);
  if (ws_size < off) return;   // not enough scratch; avoid OOB

  u16*   wcat  = (u16*)(W + o_wcat);
  u16*   wout  = (u16*)(W + o_wout);
  u16*   wcz   = (u16*)(W + o_wcz);
  float* c1    = (float*)(W + o_c1);
  float* mW    = (float*)(W + o_m);
  float* rois  = (float*)(W + o_rois);
  u16*   roisbf= (u16*)(W + o_roisbf);
  u16*   zfbf  = (u16*)(W + o_zfbf);
  float* gtp   = (float*)(W + o_gtp);
  u16*   thT   = (u16*)(W + o_thT);
  u16*   phb   = (u16*)(W + o_phb);
  u16*   gT    = (u16*)(W + o_gT);
  float* Sm    = (float*)(W + o_S);
  u16*   Pbf   = (u16*)(W + o_Pbf);
  float* Yp    = (float*)(W + o_Yp);
  u16*   Ybf   = (u16*)(W + o_Ybf);
  float* SA    = (float*)(W + o_SA);
  float* thph  = (float*)(W + o_thph);
  u16*   thbf  = (u16*)(W + o_thbf);
  u16*   phTbf = (u16*)(W + o_phTbf);
  float* fmat  = (float*)(W + o_fmat);
  float* zavg  = (float*)(W + o_zavg);
  float* zmax  = (float*)(W + o_zmax);
  float* cues  = (float*)(W + o_cues);
  float* uW    = (float*)(W + o_u);
  float* kdotW = (float*)(W + o_kdot);
  float* scW   = (float*)(W + o_sc);

  dim3 blk(256);
  auto gcx = [](int R, int C) { return dim3((unsigned)cdiv((long)R * C, 256)); };

  // ---- pack per-layer weights to bf16 ----
  k_cvt_nt<<<dim3(gcx(128,256).x,3),blk,0,stream>>>(wcat,            384*256,0, nl_g_w, 128*256,0,256, nullptr,0, 128,256,128,256, 1);
  k_cvt_nt<<<dim3(gcx(128,256).x,3),blk,0,stream>>>(wcat+128*256,    384*256,0, nl_th_w,128*256,0,256, nullptr,0, 128,256,128,256, 1);
  k_cvt_nt<<<dim3(gcx(128,256).x,3),blk,0,stream>>>(wcat+256*256,    384*256,0, nl_ph_w,128*256,0,256, nullptr,0, 128,256,128,256, 1);
  k_cvt_nt<<<dim3(gcx(256,128).x,3),blk,0,stream>>>(wout,            256*128,0, nl_out_w,256*128,0,128, nullptr,0, 256,128,256,128, 1);
  k_cvt_nt<<<dim3(gcx(256,256).x,3),blk,0,stream>>>(wcz,             512*256,0, cz_th_w,256*256,0,256, nullptr,0, 256,256,256,256, 1);
  k_cvt_nt<<<dim3(gcx(256,256).x,3),blk,0,stream>>>(wcz+256*256,     512*256,0, cz_ph_w,256*256,0,256, nullptr,0, 256,256,256,256, 1);

  // ---- mask branch, RoI pool, zf pack ----
  k_c1<<<dim3(cdiv(4*125*125,256),96),blk,0,stream>>>(c1, mask, ac1_w, ac1_b);
  k_prroi<<<dim3(96),blk,0,stream>>>(rois, zf, box);
  k_cvt_nt<<<dim3(gcx(256,240).x,96),blk,0,stream>>>(zfbf, 32L*61440,61440, zf, 32L*57600,57600,225, nullptr,0, 256,240,256,225, 32);
  k_mrest<<<dim3(96),blk,0,stream>>>(mW, c1, ac2_w, ac2_b, ac3_w, ac3_b, ac4_w, ac4_b);

  auto ggx = [](int M, int N) { return (unsigned)(((M + 63) / 64) * ((N + 31) / 32)); };
  // ---- non-local: G1 = [Wg;Wth;Wph] @ zf ----
  k_gemm_bf16<<<dim3(ggx(384,240),96),blk,0,stream>>>(wcat,384*256,0,256, zfbf,32L*61440,61440,240, gtp,32L*92160,92160,240, 384,240,256);
  k_cvt_t <<<dim3(gcx(240,128).x,96),blk,0,stream>>>(thT, 32L*30720,30720, gtp+128*240, 32L*92160,92160,240, nl_th_b,128, 240,128,225,128, 32);
  k_cvt_nt<<<dim3(gcx(128,240).x,96),blk,0,stream>>>(phb, 32L*30720,30720, gtp+256*240, 32L*92160,92160,240, nl_ph_b,128, 128,240,128,225, 32);
  k_cvt_t <<<dim3(gcx(256,128).x,96),blk,0,stream>>>(gT,  32L*32768,32768, gtp,         32L*92160,92160,240, nl_g_b, 128, 256,128,225,128, 32);
  // G2: S = th^T @ ph
  k_gemm_bf16<<<dim3(ggx(240,240),96),blk,0,stream>>>(thT,32L*30720,30720,128, phb,32L*30720,30720,240, Sm,32L*57600,57600,240, 240,240,128);
  k_softmax<<<dim3(240,96),blk,0,stream>>>(Pbf, Sm);
  // G3: Y' = P @ g^T
  k_gemm_bf16<<<dim3(ggx(240,128),96),blk,0,stream>>>(Pbf,32L*61440,61440,256, gT,32L*32768,32768,128, Yp,32L*30720,30720,128, 240,128,256);
  k_cvt_t<<<dim3(gcx(128,240).x,96),blk,0,stream>>>(Ybf, 32L*30720,30720, Yp, 32L*30720,30720,128, nullptr,0, 128,240,128,225, 32);
  // G4: SA = Wout @ Y
  k_gemm_bf16<<<dim3(ggx(256,240),96),blk,0,stream>>>(wout,256*128,0,128, Ybf,32L*30720,30720,240, SA,32L*61440,61440,240, 256,240,128);

  // ---- chan_z: G5 = [Wth;Wph] @ rois ; G6 = th @ ph^T ----
  k_cvt_nt<<<dim3(gcx(256,64).x,96),blk,0,stream>>>(roisbf, 32L*16384,16384, rois, 32L*12544,12544,49, nullptr,0, 256,64,256,49, 32);
  k_gemm_bf16<<<dim3(ggx(512,64),96),blk,0,stream>>>(wcz,512*256,0,256, roisbf,32L*16384,16384,64, thph,32L*32768,32768,64, 512,64,256);
  k_cvt_nt<<<dim3(gcx(256,64).x,96),blk,0,stream>>>(thbf,  32L*16384,16384, thph,        32L*32768,32768,64, cz_th_b,256, 256,64,256,49, 32);
  k_cvt_t <<<dim3(gcx(64,256).x,96),blk,0,stream>>>(phTbf, 32L*16384,16384, thph+256*64, 32L*32768,32768,64, cz_ph_b,256, 64,256,49,256, 32);
  k_gemm_bf16<<<dim3(ggx(256,256),96),blk,0,stream>>>(thbf,32L*16384,16384,64, phTbf,32L*16384,16384,256, fmat,32L*65536,65536,256, 256,256,64);

  // ---- cues ----
  k_pool_mlp<<<dim3(96),blk,0,stream>>>(zavg, zmax, rois, ap_w1, ap_b1, ap_w2, ap_b2, mp_w1, mp_b1, mp_w2, mp_b2);
  k_chanz_tail<<<dim3(96),blk,0,stream>>>(cues, fmat, zavg, zmax, cz_gw, cz_gb, cz_aw, cz_ab, dr_w, dr_b);

  // ---- sc branch (algebraically collapsed) ----
  k_sc_prep<<<dim3(96),blk,0,stream>>>(uW, kdotW, mW, zf, phi_w, phi_b, delta_w, delta_b);
  k_sc<<<dim3(4,96),blk,0,stream>>>(scW, xf, uW, kdotW);

  // ---- outputs ----
  const long O0 = 0;
  const long O1 = 96L * 256 * 49;          // after zf_out
  const long O2 = O1 + 96L * 256 * 225;    // after zf_full
  k_xf_out<<<dim3(cdiv(256*961,256),96),blk,0,stream>>>(out, xf, cues, scW, O2);
  k_zf_out<<<dim3(cdiv(256*225,256),96),blk,0,stream>>>(out, zf, SA, cues, bn_g, bn_b, bn_m, bn_v, O0, O1);
}
